// Model_43533788512937
// MI455X (gfx1250) — matrix operand order repaired and checked by reading, not compiled, not run on hardware
//
#include <hip/hip_runtime.h>
#include <hip/hip_bf16.h>

#define BATCH 8
#define DIM   96
#define HH    128
#define HW    (HH * HH)      // 16384
#define C4    24
#define NTOK  1024
#define HEADS 2
#define HD    12
#define QKVD  72
#define FFD   384

#define BN_INV     0.99999500003749968f   // 1/sqrt(1+1e-5)
#define ATTN_SCALE 0.28867513459481287f   // 12^-0.5

// padded row counts
#define GROWS   32     // g: 24 -> 32 channels
#define QKVROWS 224    // 7 sections * 32 rows (q h0,q h1,k h0,k h1,v h0,v h1,spill)
#define OROWS   32     // attention output / proj output rows

typedef _Float16 f16;
typedef __attribute__((ext_vector_type(16))) _Float16 v16h;
typedef __attribute__((ext_vector_type(8)))  float    v8f;

__device__ __forceinline__ v8f wmma_f16(v16h a, v16h b, v8f c) {
    return __builtin_amdgcn_wmma_f32_16x16x32_f16(false, a, false, b, (short)0, c, false, false);
}

// branch-free GELU: x * sigmoid(1.595769122*(x + 0.044715 x^3))
// native v_exp_f32 + v_rcp_f32 (denominator in (1,inf): rcp is ~1ulp here)
__device__ __forceinline__ float gelu_f(float v) {
    float u = 1.5957691216057308f * (v + 0.044715f * v * v * v);
    return v * __builtin_amdgcn_rcpf(1.0f + __expf(-u));
}

// ---- Unguarded WMMA fragment loaders (fp32 source -> fp16 fragment) ---------
// A: 16(M) x 32(K), row-major source, row stride ld. lane&15 = M row;
// lanes 0-15: K {0..15}+kBase, lanes 16-31: K {16..31}+kBase.
__device__ __forceinline__ v16h loadA(const float* __restrict__ src, int ld,
                                      int rowBase, int kBase, unsigned lane) {
    v16h a;
    const float* p = src + (size_t)(rowBase + (int)(lane & 15u)) * ld
                         + kBase + (int)((lane >> 4) << 4);
#pragma unroll
    for (int e = 0; e < 16; ++e)
        a[e] = (f16)p[e];
    return a;
}

// B: 32(K) x 16(N), channel-major source: src[k*stride + col]. lane&15 = col.
__device__ __forceinline__ v16h loadB(const float* __restrict__ src, int stride,
                                      int kBase, int colBase, unsigned lane) {
    v16h b;
    const float* p = src + (size_t)(kBase + (int)((lane >> 4) << 4)) * stride
                         + colBase + (int)(lane & 15u);
#pragma unroll
    for (int e = 0; e < 16; ++e)
        b[e] = (f16)p[(size_t)e * stride];
    return b;
}

// ---- Large GEMM kernels: one wave = 16 out-ch x 64 positions ----------------

// y = gelu(p1 @ bn0(x) + p1_b)   96->96
__global__ void k_p1(const float* __restrict__ x, const float* __restrict__ bnw,
                     const float* __restrict__ bnb, const float* __restrict__ w,
                     const float* __restrict__ bias, float* __restrict__ y) {
    unsigned lane = threadIdx.x;
    int p0 = blockIdx.x * 64, c0 = blockIdx.y * 16, b = blockIdx.z;
    int col = (int)(lane & 15u);
    const float* xb = x + (size_t)b * DIM * HW;
    v8f acc[4] = {{}, {}, {}, {}};
    for (int kc = 0; kc < DIM; kc += 32) {
        v16h a = loadA(w, DIM, c0, kc, lane);
        int kb = kc + (int)((lane >> 4) << 4);
        float sc[16], sh[16];
#pragma unroll
        for (int e = 0; e < 16; ++e) { sc[e] = bnw[kb + e] * BN_INV; sh[e] = bnb[kb + e]; }
        const float* base = xb + (size_t)kb * HW + p0 + col;
#pragma unroll
        for (int t = 0; t < 4; ++t) {
            v16h bf;
#pragma unroll
            for (int e = 0; e < 16; ++e)
                bf[e] = (f16)(base[(size_t)e * HW + t * 16] * sc[e] + sh[e]);
            acc[t] = wmma_f16(a, bf, acc[t]);
        }
    }
    int cr = c0 + (int)((lane >> 4) << 3);
    float* yb = y + (size_t)b * DIM * HW;
#pragma unroll
    for (int t = 0; t < 4; ++t)
#pragma unroll
        for (int r = 0; r < 8; ++r)
            yb[(size_t)(cr + r) * HW + p0 + t * 16 + col] = gelu_f(acc[t][r] + bias[cr + r]);
}

// xd = (la3 @ s + la3_b) * y     96->96, gated
__global__ void k_la3(const float* __restrict__ s, const float* __restrict__ w,
                      const float* __restrict__ bias, const float* __restrict__ y,
                      float* __restrict__ xd) {
    unsigned lane = threadIdx.x;
    int p0 = blockIdx.x * 64, c0 = blockIdx.y * 16, b = blockIdx.z;
    int col = (int)(lane & 15u);
    const float* sb = s + (size_t)b * DIM * HW;
    v8f acc[4] = {{}, {}, {}, {}};
    for (int kc = 0; kc < DIM; kc += 32) {
        v16h a = loadA(w, DIM, c0, kc, lane);
#pragma unroll
        for (int t = 0; t < 4; ++t) {
            v16h bf = loadB(sb, HW, kc, p0 + t * 16, lane);
            acc[t] = wmma_f16(a, bf, acc[t]);
        }
    }
    int cr = c0 + (int)((lane >> 4) << 3);
    const float* yb = y + (size_t)b * DIM * HW;
    float* ob = xd + (size_t)b * DIM * HW;
#pragma unroll
    for (int t = 0; t < 4; ++t)
#pragma unroll
        for (int r = 0; r < 8; ++r) {
            size_t o = (size_t)(cr + r) * HW + p0 + t * 16 + col;
            ob[o] = (acc[t][r] + bias[cr + r]) * yb[o];
        }
}

// z = x + (p2 @ in + p2_b) * ls1
__global__ void k_p2(const float* __restrict__ in, const float* __restrict__ w,
                     const float* __restrict__ bias, const float* __restrict__ ls,
                     const float* __restrict__ shortcut, float* __restrict__ z) {
    unsigned lane = threadIdx.x;
    int p0 = blockIdx.x * 64, c0 = blockIdx.y * 16, b = blockIdx.z;
    int col = (int)(lane & 15u);
    const float* ib = in + (size_t)b * DIM * HW;
    v8f acc[4] = {{}, {}, {}, {}};
    for (int kc = 0; kc < DIM; kc += 32) {
        v16h a = loadA(w, DIM, c0, kc, lane);
#pragma unroll
        for (int t = 0; t < 4; ++t) {
            v16h bf = loadB(ib, HW, kc, p0 + t * 16, lane);
            acc[t] = wmma_f16(a, bf, acc[t]);
        }
    }
    int cr = c0 + (int)((lane >> 4) << 3);
    const float* sb = shortcut + (size_t)b * DIM * HW;
    float* zb = z + (size_t)b * DIM * HW;
#pragma unroll
    for (int t = 0; t < 4; ++t)
#pragma unroll
        for (int r = 0; r < 8; ++r) {
            int c = cr + r;
            size_t o = (size_t)c * HW + p0 + t * 16 + col;
            zb[o] = sb[o] + (acc[t][r] + bias[c]) * ls[c];
        }
}

// f1 = fc1 @ fbn(z) + fc1_b      96->384
__global__ void k_fc1(const float* __restrict__ z, const float* __restrict__ bnw,
                      const float* __restrict__ bnb, const float* __restrict__ w,
                      const float* __restrict__ bias, float* __restrict__ out) {
    unsigned lane = threadIdx.x;
    int p0 = blockIdx.x * 64, c0 = blockIdx.y * 16, b = blockIdx.z;
    int col = (int)(lane & 15u);
    const float* zb = z + (size_t)b * DIM * HW;
    v8f acc[4] = {{}, {}, {}, {}};
    for (int kc = 0; kc < DIM; kc += 32) {
        v16h a = loadA(w, DIM, c0, kc, lane);
        int kb = kc + (int)((lane >> 4) << 4);
        float sc[16], sh[16];
#pragma unroll
        for (int e = 0; e < 16; ++e) { sc[e] = bnw[kb + e] * BN_INV; sh[e] = bnb[kb + e]; }
        const float* base = zb + (size_t)kb * HW + p0 + col;
        if (kc + 32 < DIM)
            __builtin_prefetch(base + (size_t)32 * HW, 0, 1);   // global_prefetch_b8
#pragma unroll
        for (int t = 0; t < 4; ++t) {
            v16h bf;
#pragma unroll
            for (int e = 0; e < 16; ++e)
                bf[e] = (f16)(base[(size_t)e * HW + t * 16] * sc[e] + sh[e]);
            acc[t] = wmma_f16(a, bf, acc[t]);
        }
    }
    int cr = c0 + (int)((lane >> 4) << 3);
    float* ob = out + (size_t)b * FFD * HW;
#pragma unroll
    for (int t = 0; t < 4; ++t)
#pragma unroll
        for (int r = 0; r < 8; ++r)
            ob[(size_t)(cr + r) * HW + p0 + t * 16 + col] = acc[t][r] + bias[cr + r];
}

// out = z + (fc2 @ f2 + fc2_b) * fls   384->96
__global__ void k_fc2(const float* __restrict__ f2, const float* __restrict__ w,
                      const float* __restrict__ bias, const float* __restrict__ ls,
                      const float* __restrict__ z, float* __restrict__ out) {
    unsigned lane = threadIdx.x;
    int p0 = blockIdx.x * 64, c0 = blockIdx.y * 16, b = blockIdx.z;
    int col = (int)(lane & 15u);
    const float* fb = f2 + (size_t)b * FFD * HW;
    v8f acc[4] = {{}, {}, {}, {}};
    for (int kc = 0; kc < FFD; kc += 32) {
        v16h a = loadA(w, FFD, c0, kc, lane);
        if (kc + 32 < FFD)
            __builtin_prefetch(fb + (size_t)(kc + 32 + ((lane >> 4) << 4)) * HW + p0 + col, 0, 1);
#pragma unroll
        for (int t = 0; t < 4; ++t) {
            v16h bf = loadB(fb, HW, kc, p0 + t * 16, lane);
            acc[t] = wmma_f16(a, bf, acc[t]);
        }
    }
    int cr = c0 + (int)((lane >> 4) << 3);
    const float* zb = z + (size_t)b * DIM * HW;
    float* ob = out + (size_t)b * DIM * HW;
#pragma unroll
    for (int t = 0; t < 4; ++t)
#pragma unroll
        for (int r = 0; r < 8; ++r) {
            int c = cr + r;
            size_t o = (size_t)c * HW + p0 + t * 16 + col;
            ob[o] = zb[o] + (acc[t][r] + bias[c]) * ls[c];
        }
}

// ---- Attention GEMMs (all operands zero-padded -> unguarded) ----------------

// qkv (padded layout): row (c/12)*32 + c%12  <- qkv_wp(80x32) @ g(32 rows)
__global__ void k_qkv(const float* __restrict__ g, const float* __restrict__ w,
                      float* __restrict__ qkvp) {
    unsigned lane = threadIdx.x;
    int n0 = blockIdx.x * 16, c0 = blockIdx.y * 16, b = blockIdx.z;
    v16h a  = loadA(w, 32, c0, 0, lane);
    v16h bf = loadB(g + (size_t)b * GROWS * NTOK, NTOK, 0, n0, lane);
    v8f acc = {};
    acc = wmma_f16(a, bf, acc);
    int col = n0 + (int)(lane & 15u);
    float* qb = qkvp + (size_t)b * QKVROWS * NTOK;
#pragma unroll
    for (int r = 0; r < 8; ++r) {
        int c = c0 + (int)((lane >> 4) << 3) + r;   // 0..79
        int sec = c / 12;
        int row = sec * 32 + (c - sec * 12);
        qb[(size_t)row * NTOK + col] = acc[r];
    }
}

// scores = (q*SCALE) @ k^T + rel_pos_bias
__global__ void k_scores(const float* __restrict__ qkvp, const float* __restrict__ rpb,
                         float* __restrict__ attn) {
    unsigned lane = threadIdx.x;
    int m0 = blockIdx.x * 16;      // key tile
    int q0 = blockIdx.y * 16;      // query tile
    int bh = blockIdx.z; int b = bh >> 1, h = bh & 1;
    const float* base = qkvp + (size_t)b * QKVROWS * NTOK;
    const float* qp = base + (size_t)(h * 32) * NTOK;          // q section
    const float* kp = base + (size_t)(64 + h * 32) * NTOK;     // k section
    v16h a, bf;
    {
        int q = q0 + (int)(lane & 15u);
        const float* p = qp + (size_t)((lane >> 4) << 4) * NTOK + q;
#pragma unroll
        for (int e = 0; e < 16; ++e)
            a[e] = (f16)(p[(size_t)e * NTOK] * ATTN_SCALE);
        int key = m0 + (int)(lane & 15u);
        const float* pk = kp + (size_t)((lane >> 4) << 4) * NTOK + key;
#pragma unroll
        for (int e = 0; e < 16; ++e)
            bf[e] = (f16)pk[(size_t)e * NTOK];
    }
    v8f acc = {};
    acc = wmma_f16(a, bf, acc);
    int key = m0 + (int)(lane & 15u);
    int ky = key >> 5, kx = key & 31;
    float* ap = attn + (size_t)bh * NTOK * NTOK;
#pragma unroll
    for (int r = 0; r < 8; ++r) {
        int q  = q0 + (int)((lane >> 4) << 3) + r;
        int qy = q >> 5, qx = q & 31;
        int ridx = (qy - ky + 31) * 63 + (qx - kx + 31);
        ap[(size_t)q * NTOK + key] = acc[r] + rpb[ridx * 2 + h];
    }
}

__global__ void k_softmax(float* __restrict__ attn) {
    int row = blockIdx.x;
    float* p = attn + (size_t)row * NTOK;
    __shared__ float red[128];
    int t = threadIdx.x;
    float mx = -3.4e38f;
    for (int i = t; i < NTOK; i += 128) mx = fmaxf(mx, p[i]);
    red[t] = mx; __syncthreads();
    for (int s = 64; s > 0; s >>= 1) { if (t < s) red[t] = fmaxf(red[t], red[t + s]); __syncthreads(); }
    mx = red[0]; __syncthreads();
    float sum = 0.0f;
    for (int i = t; i < NTOK; i += 128) { float e = __expf(p[i] - mx); p[i] = e; sum += e; }
    red[t] = sum; __syncthreads();
    for (int s = 64; s > 0; s >>= 1) { if (t < s) red[t] += red[t + s]; __syncthreads(); }
    float inv = __builtin_amdgcn_rcpf(red[0]);
    for (int i = t; i < NTOK; i += 128) p[i] *= inv;
}

// o = attn @ v ; store padded rows h*16+d (d>=12 results are exactly 0)
__global__ void k_av(const float* __restrict__ attn, const float* __restrict__ qkvp,
                     float* __restrict__ obuf) {
    unsigned lane = threadIdx.x;
    int q0 = blockIdx.x * 16;
    int bh = blockIdx.z; int b = bh >> 1, h = bh & 1;
    const float* ap = attn + (size_t)bh * NTOK * NTOK;
    const float* vp = qkvp + (size_t)b * QKVROWS * NTOK + (size_t)(128 + h * 32) * NTOK;
    int d = (int)(lane & 15u);
    const float* vrow = vp + (size_t)d * NTOK;
    v8f acc = {};
    for (int kc = 0; kc < NTOK; kc += 32) {
        v16h a = loadA(ap, NTOK, q0, kc, lane);
        v16h bf;
        const float* pv = vrow + kc + (int)((lane >> 4) << 4);  // 16 consecutive keys
#pragma unroll
        for (int e = 0; e < 16; ++e) bf[e] = (f16)pv[e];
        acc = wmma_f16(a, bf, acc);
    }
    float* op = obuf + ((size_t)b * OROWS + h * 16 + d) * NTOK;
#pragma unroll
    for (int r = 0; r < 8; ++r)
        op[q0 + (int)((lane >> 4) << 3) + r] = acc[r];
}

// po = pr_wp(32x32, col-remapped) @ o + pr_bp
__global__ void k_proj(const float* __restrict__ obuf, const float* __restrict__ w,
                       const float* __restrict__ bias, float* __restrict__ po) {
    unsigned lane = threadIdx.x;
    int n0 = blockIdx.x * 16, c0 = blockIdx.y * 16, b = blockIdx.z;
    v16h a  = loadA(w, 32, c0, 0, lane);
    v16h bf = loadB(obuf + (size_t)b * OROWS * NTOK, NTOK, 0, n0, lane);
    v8f acc = {};
    acc = wmma_f16(a, bf, acc);
    int col = n0 + (int)(lane & 15u);
    float* pb = po + (size_t)b * OROWS * NTOK;
#pragma unroll
    for (int r = 0; r < 8; ++r) {
        int c = c0 + (int)((lane >> 4) << 3) + r;
        pb[(size_t)c * NTOK + col] = acc[r] + bias[c];
    }
}

// xo = co_wp(96x32) @ po + co_b
__global__ void k_co(const float* __restrict__ po, const float* __restrict__ w,
                     const float* __restrict__ bias, float* __restrict__ xo) {
    unsigned lane = threadIdx.x;
    int n0 = blockIdx.x * 16, c0 = blockIdx.y * 16, b = blockIdx.z;
    v16h a  = loadA(w, 32, c0, 0, lane);
    v16h bf = loadB(po + (size_t)b * OROWS * NTOK, NTOK, 0, n0, lane);
    v8f acc = {};
    acc = wmma_f16(a, bf, acc);
    int col = n0 + (int)(lane & 15u);
    float* xb = xo + (size_t)b * DIM * NTOK;
#pragma unroll
    for (int r = 0; r < 8; ++r) {
        int c = c0 + (int)((lane >> 4) << 3) + r;
        xb[(size_t)c * NTOK + col] = acc[r] + bias[c];
    }
}

// ---- Elementwise / depthwise kernels ---------------------------------------

__global__ void k_fill0(float* __restrict__ p, int n) {
    int i = blockIdx.x * blockDim.x + threadIdx.x;
    if (i < n) p[i] = 0.0f;
}

// pad/copy small weights into zero-initialized padded buffers
__global__ void k_prep(const float* __restrict__ qkvw, const float* __restrict__ prw,
                       const float* __restrict__ prb, const float* __restrict__ cow,
                       float* __restrict__ qkvwp, float* __restrict__ prwp,
                       float* __restrict__ prbp, float* __restrict__ cowp) {
    int i = blockIdx.x * blockDim.x + threadIdx.x;
    if (i < QKVD * C4) {                                    // qkv_w 72x24 -> 80x32
        int r = i / C4, c = i % C4;
        qkvwp[r * 32 + c] = qkvw[i];
    } else if (i < QKVD * C4 + C4 * C4) {                   // pr_w 24x24 -> 32x32 remap cols
        int j = i - QKVD * C4; int r = j / C4, c = j % C4;
        prwp[r * 32 + (c / HD) * 16 + (c % HD)] = prw[j];
    } else if (i < QKVD * C4 + C4 * C4 + DIM * C4) {        // co_w 96x24 -> 96x32
        int j = i - QKVD * C4 - C4 * C4; int r = j / C4, c = j % C4;
        cowp[r * 32 + c] = cow[j];
    } else if (i < QKVD * C4 + C4 * C4 + DIM * C4 + C4) {   // pr_b 24 -> 32
        int j = i - QKVD * C4 - C4 * C4 - DIM * C4;
        prbp[j] = prb[j];
    }
}

__global__ void k_dw3x3_set(const float* __restrict__ in, const float* __restrict__ w,
                            const float* __restrict__ bias, float* __restrict__ out) {
    int idx = blockIdx.x * blockDim.x + threadIdx.x;
    if (idx >= BATCH * DIM * HW) return;
    int p = idx & (HW - 1);
    int c = (idx >> 14) % DIM;
    int hx = p & 127, hy = p >> 7;
    const float* ip = in + (idx - p);
    const float* wp = w + c * 9;
    float s = bias[c];
#pragma unroll
    for (int dy = -1; dy <= 1; ++dy)
#pragma unroll
        for (int dx = -1; dx <= 1; ++dx) {
            int yy = hy + dy, xx = hx + dx;
            if (yy >= 0 && yy < HH && xx >= 0 && xx < HH)
                s += wp[(dy + 1) * 3 + (dx + 1)] * ip[yy * HH + xx];
        }
    out[idx] = s;
}

__global__ void k_dwh(const float* __restrict__ in, const float* __restrict__ w,
                      const float* __restrict__ bias, float* __restrict__ out, int T) {
    int idx = blockIdx.x * blockDim.x + threadIdx.x;
    if (idx >= BATCH * DIM * HW) return;
    int p = idx & (HW - 1);
    int c = (idx >> 14) % DIM;
    int hx = p & 127;
    const float* ip = in + (idx - hx);
    const float* wp = w + c * T;
    int half = T >> 1;
    float s = bias[c];
    for (int i = 0; i < T; ++i) {
        int xx = hx + i - half;
        if (xx >= 0 && xx < HH) s += wp[i] * ip[xx];
    }
    out[idx] = s;
}

__global__ void k_dwv_acc(const float* __restrict__ in, const float* __restrict__ w,
                          const float* __restrict__ bias, float* __restrict__ out, int T) {
    int idx = blockIdx.x * blockDim.x + threadIdx.x;
    if (idx >= BATCH * DIM * HW) return;
    int p = idx & (HW - 1);
    int c = (idx >> 14) % DIM;
    int hx = p & 127, hy = p >> 7;
    const float* ip = in + (idx - p);
    const float* wp = w + c * T;
    int half = T >> 1;
    float s = bias[c];
    for (int i = 0; i < T; ++i) {
        int yy = hy + i - half;
        if (yy >= 0 && yy < HH) s += wp[i] * ip[yy * HH + hx];
    }
    out[idx] += s;
}

__global__ void k_dw3x3_gelu(const float* __restrict__ in, const float* __restrict__ w,
                             const float* __restrict__ bias, float* __restrict__ out) {
    int idx = blockIdx.x * blockDim.x + threadIdx.x;
    if (idx >= BATCH * FFD * HW) return;
    int p = idx & (HW - 1);
    int c = (idx >> 14) % FFD;
    int hx = p & 127, hy = p >> 7;
    const float* ip = in + (idx - p);
    const float* wp = w + c * 9;
    float s = bias[c];
#pragma unroll
    for (int dy = -1; dy <= 1; ++dy)
#pragma unroll
        for (int dx = -1; dx <= 1; ++dx) {
            int yy = hy + dy, xx = hx + dx;
            if (yy >= 0 && yy < HH && xx >= 0 && xx < HH)
                s += wp[(dy + 1) * 3 + (dx + 1)] * ip[yy * HH + xx];
        }
    out[idx] = gelu_f(s);
}

// grouped 4x4 stride-4 conv + BN + ReLU6 -> g rows 0..23 of padded 32
__global__ void k_ge(const float* __restrict__ y, const float* __restrict__ w,
                     const float* __restrict__ bias, const float* __restrict__ gw,
                     const float* __restrict__ gb, float* __restrict__ g) {
    int idx = blockIdx.x * blockDim.x + threadIdx.x;
    if (idx >= BATCH * C4 * NTOK) return;
    int n = idx & (NTOK - 1);
    int j = n & 31, i = n >> 5;
    int c = (idx >> 10) % C4, b = idx / (C4 * NTOK);
    const float* yb = y + ((size_t)b * DIM + c * 4) * HW;
    const float* wp = w + c * 64;
    float s = bias[c];
#pragma unroll
    for (int ci = 0; ci < 4; ++ci)
#pragma unroll
        for (int kh = 0; kh < 4; ++kh)
#pragma unroll
            for (int kw = 0; kw < 4; ++kw)
                s += wp[ci * 16 + kh * 4 + kw] *
                     yb[(size_t)ci * HW + (4 * i + kh) * HH + (4 * j + kw)];
    s = s * (gw[c] * BN_INV) + gb[c];
    g[((size_t)b * GROWS + c) * NTOK + n] = fminf(fmaxf(s, 0.0f), 6.0f);
}

// bilinear x4 upsample (align_corners) of 32x32 + add xd
__global__ void k_up_add(const float* __restrict__ xo, const float* __restrict__ xd,
                         float* __restrict__ out) {
    int idx = blockIdx.x * blockDim.x + threadIdx.x;
    if (idx >= BATCH * DIM * HW) return;
    int p = idx & (HW - 1);
    int hx = p & 127, hy = p >> 7;
    const float* sp = xo + (size_t)(idx >> 14) * NTOK;
    const float RS = 31.0f / 127.0f;
    float fy = hy * RS; int y0 = (int)fy; float ty = fy - y0; int y1 = min(y0 + 1, 31);
    float fx = hx * RS; int x0 = (int)fx; float tx = fx - x0; int x1 = min(x0 + 1, 31);
    float v = (sp[y0 * 32 + x0] * (1.0f - ty) + sp[y1 * 32 + x0] * ty) * (1.0f - tx)
            + (sp[y0 * 32 + x1] * (1.0f - ty) + sp[y1 * 32 + x1] * ty) * tx;
    out[idx] = v + xd[idx];
}

// ---- Host launch ------------------------------------------------------------

extern "C" void kernel_launch(void* const* d_in, const int* in_sizes, int n_in,
                              void* d_out, int out_size, void* d_ws, size_t ws_size,
                              hipStream_t stream) {
    (void)in_sizes; (void)n_in; (void)out_size; (void)ws_size;
    const float* x     = (const float*)d_in[0];
    const float* bn0w  = (const float*)d_in[1];
    const float* bn0b  = (const float*)d_in[2];
    const float* p1w   = (const float*)d_in[3];
    const float* p1b   = (const float*)d_in[4];
    const float* la0w  = (const float*)d_in[5];
    const float* la0b  = (const float*)d_in[6];
    const float* la01w = (const float*)d_in[7];
    const float* la01b = (const float*)d_in[8];
    const float* la02w = (const float*)d_in[9];
    const float* la02b = (const float*)d_in[10];
    const float* la11w = (const float*)d_in[11];
    const float* la11b = (const float*)d_in[12];
    const float* la12w = (const float*)d_in[13];
    const float* la12b = (const float*)d_in[14];
    const float* la3w  = (const float*)d_in[15];
    const float* la3b  = (const float*)d_in[16];
    const float* gew   = (const float*)d_in[17];
    const float* geb   = (const float*)d_in[18];
    const float* gbnw  = (const float*)d_in[19];
    const float* gbnb  = (const float*)d_in[20];
    const float* rpb   = (const float*)d_in[21];
    const float* qkvw  = (const float*)d_in[22];
    const float* prw   = (const float*)d_in[23];
    const float* prb   = (const float*)d_in[24];
    const float* cow   = (const float*)d_in[25];
    const float* cob   = (const float*)d_in[26];
    const float* p2w   = (const float*)d_in[27];
    const float* p2b   = (const float*)d_in[28];
    const float* ls1   = (const float*)d_in[29];
    const float* fbnw  = (const float*)d_in[30];
    const float* fbnb  = (const float*)d_in[31];
    const float* fc1w  = (const float*)d_in[32];
    const float* fc1b  = (const float*)d_in[33];
    const float* dww   = (const float*)d_in[34];
    const float* dwb   = (const float*)d_in[35];
    const float* fc2w  = (const float*)d_in[36];
    const float* fc2b  = (const float*)d_in[37];
    const float* fls   = (const float*)d_in[38];
    float* out = (float*)d_out;

    float* ws = (float*)d_ws;
    const size_t S = (size_t)BATCH * DIM * HW;   // 12.58M floats
    float* y    = ws;
    float* bufA = ws + S;          // separable tmp, then xd
    float* bufB = ws + 2 * S;      // dw sum, then xd+xg
    float* bufZ = ws + 3 * S;      // z
    float* f1   = ws + 4 * S;      // fc1 out (4S); attn aliases this earlier
    float* f2   = ws + 8 * S;      // dw+gelu out (4S)
    float* attn = f1;              // 16.7M floats << 4S

    float* smallb = ws + 12 * S;
    float* gp    = smallb;                                   // 8*32*1024
    float* qkvp  = gp    + (size_t)BATCH * GROWS  * NTOK;    // 8*224*1024
    float* obuf  = qkvp  + (size_t)BATCH * QKVROWS* NTOK;    // 8*32*1024
    float* pob   = obuf  + (size_t)BATCH * OROWS  * NTOK;    // 8*32*1024
    float* xo    = pob   + (size_t)BATCH * OROWS  * NTOK;    // 8*96*1024
    float* qkvwp = xo    + (size_t)BATCH * DIM    * NTOK;    // 80*32
    float* prwp  = qkvwp + 80 * 32;                          // 32*32
    float* cowp  = prwp  + 32 * 32;                          // 96*32
    float* prbp  = cowp  + 96 * 32;                          // 32
    const int smallN = (int)((prbp + 32) - smallb);

    const int EL  = BATCH * DIM * HW;
    const int EL4 = BATCH * FFD * HW;

    // zero padded staging region, then build padded weights
    k_fill0<<<(smallN + 255) / 256, 256, 0, stream>>>(smallb, smallN);
    k_prep<<<(QKVD * C4 + C4 * C4 + DIM * C4 + C4 + 255) / 256, 256, 0, stream>>>(
        qkvw, prw, prb, cow, qkvwp, prwp, prbp, cowp);

    // y = gelu(p1 @ bn0(x))
    k_p1<<<dim3(HW / 64, DIM / 16, BATCH), 32, 0, stream>>>(x, bn0w, bn0b, p1w, p1b, y);

    // LAP
    k_dw3x3_set<<<(EL + 255) / 256, 256, 0, stream>>>(y, la0w, la0b, bufB);
    k_dwh<<<(EL + 255) / 256, 256, 0, stream>>>(y, la01w, la01b, bufA, 5);
    k_dwv_acc<<<(EL + 255) / 256, 256, 0, stream>>>(bufA, la02w, la02b, bufB, 5);
    k_dwh<<<(EL + 255) / 256, 256, 0, stream>>>(y, la11w, la11b, bufA, 11);
    k_dwv_acc<<<(EL + 255) / 256, 256, 0, stream>>>(bufA, la12w, la12b, bufB, 11);
    k_la3<<<dim3(HW / 64, DIM / 16, BATCH), 32, 0, stream>>>(bufB, la3w, la3b, y, bufA);

    // GAP
    k_ge<<<(BATCH * C4 * NTOK + 255) / 256, 256, 0, stream>>>(y, gew, geb, gbnw, gbnb, gp);
    k_qkv<<<dim3(NTOK / 16, 5, BATCH), 32, 0, stream>>>(gp, qkvwp, qkvp);
    k_scores<<<dim3(NTOK / 16, NTOK / 16, BATCH * HEADS), 32, 0, stream>>>(qkvp, rpb, attn);
    k_softmax<<<BATCH * HEADS * NTOK, 128, 0, stream>>>(attn);
    k_av<<<dim3(NTOK / 16, 1, BATCH * HEADS), 32, 0, stream>>>(attn, qkvp, obuf);
    k_proj<<<dim3(NTOK / 16, 2, BATCH), 32, 0, stream>>>(obuf, prwp, prbp, pob);
    k_co<<<dim3(NTOK / 16, DIM / 16, BATCH), 32, 0, stream>>>(pob, cowp, cob, xo);
    k_up_add<<<(EL + 255) / 256, 256, 0, stream>>>(xo, bufA, bufB);

    // merge
    k_p2<<<dim3(HW / 64, DIM / 16, BATCH), 32, 0, stream>>>(bufB, p2w, p2b, ls1, x, bufZ);

    // FFN
    k_fc1<<<dim3(HW / 64, FFD / 16, BATCH), 32, 0, stream>>>(bufZ, fbnw, fbnb, fc1w, fc1b, f1);
    k_dw3x3_gelu<<<(EL4 + 255) / 256, 256, 0, stream>>>(f1, dww, dwb, f2);
    k_fc2<<<dim3(HW / 64, DIM / 16, BATCH), 32, 0, stream>>>(f2, fc2w, fc2b, fls, bufZ, out);
}
